// Baseline_cbr_mb_2_38757784879353
// MI455X (gfx1250) — matrix operand order repaired and checked by reading, not compiled, not run on hardware
//
#include <hip/hip_runtime.h>
#include <hip/hip_bf16.h>

typedef __attribute__((ext_vector_type(16))) _Float16 v16h;
typedef __attribute__((ext_vector_type(8)))  float    v8f;

#define NF 50000
#define NL 5000
#define LP 8
#define PP 80
#define NITERS 12

// ---- branch-free fast math: v_exp_f32 / v_rcp_f32, no EXEC divergence ----
__device__ __forceinline__ float rcp_f(float x){ return __builtin_amdgcn_rcpf(x); }
__device__ __forceinline__ float sigmoid_f(float x){ return rcp_f(1.f + __expf(-x)); }
__device__ __forceinline__ float tanh_f(float x){
  float xc = fminf(fmaxf(x, -15.f), 15.f);       // tanh saturated far before +-15
  float e  = __expf(2.f * xc);
  return (e - 1.f) * rcp_f(e + 1.f);
}
__device__ __forceinline__ float selu_f(float x){
  const float scale = 1.0507009873554805f, alpha = 1.6732632423543772f;
  return x > 0.f ? scale * x : scale * alpha * (__expf(x) - 1.f);
}
__device__ __forceinline__ float softplus_f(float x){
  float r = __logf(1.f + __expf(fminf(x, 20.f)));
  return x > 20.f ? x : r;                        // lowers to v_cndmask (no branch)
}

// ---------------- flow encoder: feats(13) -> 16 -> 16, SELU ----------------
__global__ __launch_bounds__(256) void flow_encode_kernel(
    const float* __restrict__ traffic, const float* __restrict__ packets,
    const float* __restrict__ pkts_pb, const float* __restrict__ bitrate_pb,
    const float* __restrict__ pkt_size, const float* __restrict__ ftype,
    const float* __restrict__ ipg_mean, const float* __restrict__ ibg,
    const float* __restrict__ p90, const float* __restrict__ rate,
    const float* __restrict__ ipg_var, const int* __restrict__ flen,
    const float* __restrict__ W1, const float* __restrict__ b1,
    const float* __restrict__ W2, const float* __restrict__ b2,
    float* __restrict__ ps)
{
  __shared__ float sW1[13*16], sW2[16*16], sb1[16], sb2[16];
  for (int i = threadIdx.x; i < 13*16; i += blockDim.x) sW1[i] = W1[i];
  for (int i = threadIdx.x; i < 16*16; i += blockDim.x) sW2[i] = W2[i];
  if (threadIdx.x < 16){ sb1[threadIdx.x] = b1[threadIdx.x]; sb2[threadIdx.x] = b2[threadIdx.x]; }
  __syncthreads();
  int f = blockIdx.x * blockDim.x + threadIdx.x;
  if (f >= NF) return;
  #define NRMV(x) (((x) - 0.1f) * 2.0f)
  float ft[13];
  ft[0]  = NRMV(traffic[f]);   ft[1]  = NRMV(packets[f]);
  ft[2]  = NRMV(ibg[f]);       ft[3]  = NRMV(rate[f]);
  ft[4]  = NRMV(p90[f]);       ft[5]  = NRMV(pkt_size[f]);
  ft[6]  = NRMV(bitrate_pb[f]);ft[7]  = NRMV(ipg_mean[f]);
  ft[8]  = NRMV(ipg_var[f]);   ft[9]  = NRMV(pkts_pb[f]);
  ft[10] = (float)flen[f];
  ft[11] = ftype[f*2+0]; ft[12] = ftype[f*2+1];
  float h1[16];
  #pragma unroll
  for (int j = 0; j < 16; j++){
    float s = sb1[j];
    #pragma unroll
    for (int k = 0; k < 13; k++) s += ft[k] * sW1[k*16 + j];
    h1[j] = selu_f(s);
  }
  #pragma unroll
  for (int j = 0; j < 16; j++){
    float s = sb2[j];
    #pragma unroll
    for (int k = 0; k < 16; k++) s += h1[k] * sW2[k*16 + j];
    ps[f*16 + j] = selu_f(s);
  }
}

// ---------------- link encoder: load + feats(3) -> 16 -> 16, SELU ----------------
__global__ __launch_bounds__(256) void link_encode_kernel(
    const float* __restrict__ traffic, const float* __restrict__ cap,
    const float* __restrict__ maxload, const int* __restrict__ ptl,
    const float* __restrict__ W1, const float* __restrict__ b1,
    const float* __restrict__ W2, const float* __restrict__ b2,
    float* __restrict__ lstate)
{
  int l = blockIdx.x * blockDim.x + threadIdx.x;
  if (l >= NL) return;
  float s = 0.f;
  for (int p = 0; p < PP; p++) s += traffic[ptl[(l*PP + p)*2]];
  float load  = s / (cap[l] * 1e9f);
  float nload = load / maxload[0];
  float lf[3] = { (cap[l] - 0.1f) * 2.0f, load, nload };
  float h1[16];
  #pragma unroll
  for (int j = 0; j < 16; j++){
    float a = b1[j];
    #pragma unroll
    for (int k = 0; k < 3; k++) a += lf[k] * W1[k*16 + j];
    h1[j] = selu_f(a);
  }
  #pragma unroll
  for (int j = 0; j < 16; j++){
    float a = b2[j];
    #pragma unroll
    for (int k = 0; k < 16; k++) a += h1[k] * W2[k*16 + j];
    lstate[l*16 + j] = selu_f(a);
  }
}

// ---------------- path GRU scan: 8 steps, fused [x|h] K=32 WMMA ----------------
// wave handles 16 flows; A = [x | h] (16x32 f16); B gate tiles = [Wk_g ; Wr_g] (32x16 f16)
__global__ __launch_bounds__(256) void path_gru_wmma_kernel(
    const float* __restrict__ lstate,   // (NL,16)
    float* __restrict__ ps,             // (NF,16) in/out
    float* __restrict__ seqb,           // (NF,9,16)
    const int* __restrict__ ltp,        // (NF,8)
    const float* __restrict__ Wk,       // (16,48)
    const float* __restrict__ Wr,       // (16,48)
    const float* __restrict__ bias)     // (2,48)
{
  __shared__ float lds_h[8][16][17];    // per-wave 16x16 transpose tile (padded)
  int wave = threadIdx.x >> 5;
  int lane = threadIdx.x & 31;
  int lh   = lane >> 4;                 // lane half: 0/1
  int ln   = lane & 15;
  int flow_base = (blockIdx.x * 8 + wave) * 16;
  if (flow_base > NF - 16) flow_base = NF - 16;   // keep EXEC all-ones; dup writes benign

  // B layout (32x16 f16): element k<8 = Wk row lh*8+k (K=lh*8+k), element k+8 = Wr row lh*8+k (K=16+lh*8+k)
  v16h B0, B1, B2a, B2b;
  #pragma unroll
  for (int k = 0; k < 8; k++){
    int kr = lh*8 + k;
    B0[k]  = (_Float16)Wk[kr*48 + 0  + ln];  B0[k + 8] = (_Float16)Wr[kr*48 + 0  + ln];
    B1[k]  = (_Float16)Wk[kr*48 + 16 + ln];  B1[k + 8] = (_Float16)Wr[kr*48 + 16 + ln];
    B2a[k] = (_Float16)Wk[kr*48 + 32 + ln];  B2a[k + 8] = (_Float16)0.f;   // ah only from x
    B2b[k] = (_Float16)0.f;  B2b[k + 8] = (_Float16)Wr[kr*48 + 32 + ln];   // gh only from h
  }
  float bz  = bias[0  + ln] + bias[48 + ln];   // combined z bias
  float br  = bias[16 + ln] + bias[64 + ln];   // combined r bias
  float b0h = bias[32 + ln], b1h = bias[80 + ln];

  // h in WMMA C layout: element (m = r + 8*lh, n = ln); write seq slot 0 (prev state)
  float h[8];
  #pragma unroll
  for (int r = 0; r < 8; r++){
    int flow = flow_base + r + lh*8;
    h[r] = ps[flow*16 + ln];
    seqb[flow*144 + ln] = h[r];
  }
  int flowA = flow_base + ln;           // A-operand row for this lane (M = ln)

  for (int t = 0; t < LP; t++){
    // stage h (C layout) into LDS -> read back in A layout (same-wave DS ops are in order)
    #pragma unroll
    for (int r = 0; r < 8; r++) lds_h[wave][r + lh*8][ln] = h[r];
    int li = ltp[flowA*LP + t];
    const float* xr = lstate + li*16 + lh*8;
    v16h A;
    #pragma unroll
    for (int kk = 0; kk < 8; kk++){
      A[kk]     = (_Float16)xr[kk];                       // K = lh*8+kk        (x part)
      A[kk + 8] = (_Float16)lds_h[wave][ln][lh*8 + kk];   // K = 16 + lh*8+kk   (h part)
    }
    v8f c0 = {}, c1 = {}, c2 = {}, c3 = {};
    c0 = __builtin_amdgcn_wmma_f32_16x16x32_f16(false, A, false, B0,  (short)0, c0, false, false); // az+gz
    c1 = __builtin_amdgcn_wmma_f32_16x16x32_f16(false, A, false, B1,  (short)0, c1, false, false); // ar+gr
    c2 = __builtin_amdgcn_wmma_f32_16x16x32_f16(false, A, false, B2a, (short)0, c2, false, false); // ah
    c3 = __builtin_amdgcn_wmma_f32_16x16x32_f16(false, A, false, B2b, (short)0, c3, false, false); // gh
    #pragma unroll
    for (int r = 0; r < 8; r++){
      float z    = sigmoid_f(c0[r] + bz);
      float rr   = sigmoid_f(c1[r] + br);
      float cand = tanh_f((c2[r] + b0h) + rr * (c3[r] + b1h));
      h[r] = z * h[r] + (1.f - z) * cand;
      int flow = flow_base + r + lh*8;
      seqb[flow*144 + (t + 1)*16 + ln] = h[r];
    }
  }
  #pragma unroll
  for (int r = 0; r < 8; r++){
    int flow = flow_base + r + lh*8;
    ps[flow*16 + ln] = h[r];
  }
}

// ---------------- attention + weighted path sum per link (WMMA 80x16 @ 16x16) ----------------
__global__ __launch_bounds__(256) void attention_wmma_kernel(
    const float* __restrict__ seqb, const int* __restrict__ ptl,
    const float* __restrict__ attW, const float* __restrict__ attb,
    float* __restrict__ psum)
{
  int wave = threadIdx.x >> 5;
  int lane = threadIdx.x & 31;
  int lh   = lane >> 4;
  int ln   = lane & 15;
  int link = blockIdx.x * 8 + wave;     // grid = NL/8 = 625 exact

  v16h B;
  #pragma unroll
  for (int k = 0; k < 16; k++)
    B[k] = k >= 8 ? (_Float16)0.f : (_Float16)attW[(lh*8 + k)*16 + ln];  // K=lh*8+k real for k<8, K16..31 padded
  float ab  = attb[ln];
  float acc = 0.f;

  for (int s = 0; s < 5; s++){          // 5 tiles of 16 rows cover P=80
    int p  = 16*s + ln;
    int fi = ptl[(link*PP + p)*2 + 0];
    int po = ptl[(link*PP + p)*2 + 1];
    const float* row = seqb + (fi*9 + po)*16 + lh*8;
    v16h A;
    #pragma unroll
    for (int kk = 0; kk < 8; kk++){ A[kk] = (_Float16)row[kk]; A[kk + 8] = (_Float16)0.f; }
    v8f c = {};
    c = __builtin_amdgcn_wmma_f32_16x16x32_f16(false, A, false, B, (short)0, c, false, false);
    #pragma unroll
    for (int r = 0; r < 8; r++){
      float att = c[r] + ab;
      att = att > 0.f ? att : 0.01f * att;          // leaky_relu (v_cndmask)
      float m = att;                                 // row max over n (16 lanes in half)
      #pragma unroll
      for (int d = 1; d < 16; d <<= 1) m = fmaxf(m, __shfl_xor(m, d, 32));
      float e = __expf(att - m);
      float ssum = e;
      #pragma unroll
      for (int d = 1; d < 16; d <<= 1) ssum += __shfl_xor(ssum, d, 32);
      float score = e * rcp_f(ssum);
      int p2 = 16*s + r + lh*8;                      // C-layout row of this element
      int f2 = ptl[(link*PP + p2)*2 + 0];
      int o2 = ptl[(link*PP + p2)*2 + 1];
      float pg = seqb[(f2*9 + o2)*16 + ln];
      acc += score * pg;
    }
  }
  acc += __shfl_xor(acc, 16, 32);                    // combine halves (m blocks)
  if (lh == 0) psum[link*16 + ln] = acc;
}

// ---------------- link GRU (5000 links, trivial FLOPs) ----------------
__global__ __launch_bounds__(256) void link_gru_kernel(
    float* __restrict__ lstate, const float* __restrict__ psum,
    const float* __restrict__ Wk, const float* __restrict__ Wr,
    const float* __restrict__ bias)
{
  __shared__ float sWk[768], sWr[768], sb[96];
  for (int i = threadIdx.x; i < 768; i += blockDim.x){ sWk[i] = Wk[i]; sWr[i] = Wr[i]; }
  for (int i = threadIdx.x; i < 96;  i += blockDim.x) sb[i] = bias[i];
  __syncthreads();
  int l = blockIdx.x * blockDim.x + threadIdx.x;
  if (l >= NL) return;
  float x[16], h[16];
  #pragma unroll
  for (int k = 0; k < 16; k++){ x[k] = psum[l*16 + k]; h[k] = lstate[l*16 + k]; }
  float a[48], g[48];
  for (int j = 0; j < 48; j++){
    float sa = sb[j], sg = sb[48 + j];
    #pragma unroll
    for (int k = 0; k < 16; k++){ sa += x[k] * sWk[k*48 + j]; sg += h[k] * sWr[k*48 + j]; }
    a[j] = sa; g[j] = sg;
  }
  #pragma unroll
  for (int d = 0; d < 16; d++){
    float z = sigmoid_f(a[d] + g[d]);
    float r = sigmoid_f(a[16 + d] + g[16 + d]);
    lstate[l*16 + d] = z * h[d] + (1.f - z) * tanh_f(a[32 + d] + r * g[32 + d]);
  }
}

// ---------------- readout: seq[:,1:] -> 8 -> 4 -> softplus -> sum(occ/cap) ----------------
__global__ __launch_bounds__(256) void readout_kernel(
    const float* __restrict__ seqb, const int* __restrict__ ltp,
    const float* __restrict__ cap,
    const float* __restrict__ W1, const float* __restrict__ b1,
    const float* __restrict__ W2, const float* __restrict__ b2,
    const float* __restrict__ W3, const float* __restrict__ b3,
    float* __restrict__ out)
{
  __shared__ float sW1[128], sW2[32], sW3[4], sb1[8], sb2[4], sb3[1];
  for (int i = threadIdx.x; i < 128; i += blockDim.x) sW1[i] = W1[i];
  if (threadIdx.x < 32) sW2[threadIdx.x] = W2[threadIdx.x];
  if (threadIdx.x < 8)  sb1[threadIdx.x] = b1[threadIdx.x];
  if (threadIdx.x < 4){ sW3[threadIdx.x] = W3[threadIdx.x]; sb2[threadIdx.x] = b2[threadIdx.x]; }
  if (threadIdx.x == 0) sb3[0] = b3[0];
  __syncthreads();
  int f = blockIdx.x * blockDim.x + threadIdx.x;
  if (f >= NF) return;
  float qd = 0.f;
  for (int t = 0; t < LP; t++){
    const float* v = seqb + f*144 + (t + 1)*16;
    float h1[8];
    #pragma unroll
    for (int j = 0; j < 8; j++){
      float s = sb1[j];
      #pragma unroll
      for (int k = 0; k < 16; k++) s += v[k] * sW1[k*8 + j];
      h1[j] = selu_f(s);
    }
    float h2[4];
    #pragma unroll
    for (int j = 0; j < 4; j++){
      float s = sb2[j];
      #pragma unroll
      for (int k = 0; k < 8; k++) s += h1[k] * sW2[k*4 + j];
      h2[j] = selu_f(s);
    }
    float o = sb3[0];
    #pragma unroll
    for (int k = 0; k < 4; k++) o += h2[k] * sW3[k];
    o = softplus_f(o);
    qd += o * rcp_f(cap[ltp[f*LP + t]]);
  }
  out[f] = qd;
}

extern "C" void kernel_launch(void* const* d_in, const int* in_sizes, int n_in,
                              void* d_out, int out_size, void* d_ws, size_t ws_size,
                              hipStream_t stream)
{
  const float* flow_traffic    = (const float*)d_in[0];
  const float* flow_packets    = (const float*)d_in[1];
  const float* max_link_load   = (const float*)d_in[2];
  const float* flow_pkts_pb    = (const float*)d_in[3];
  const float* flow_bitrate_pb = (const float*)d_in[4];
  const float* flow_pkt_size   = (const float*)d_in[5];
  const float* flow_type       = (const float*)d_in[6];
  const float* flow_ipg_mean   = (const float*)d_in[7];
  const float* ibg             = (const float*)d_in[8];
  const float* flow_p90        = (const float*)d_in[9];
  const float* rate            = (const float*)d_in[10];
  const float* flow_ipg_var    = (const float*)d_in[11];
  const float* link_capacity   = (const float*)d_in[12];
  const float* fe_W1 = (const float*)d_in[13];
  const float* fe_b1 = (const float*)d_in[14];
  const float* fe_W2 = (const float*)d_in[15];
  const float* fe_b2 = (const float*)d_in[16];
  const float* le_W1 = (const float*)d_in[17];
  const float* le_b1 = (const float*)d_in[18];
  const float* le_W2 = (const float*)d_in[19];
  const float* le_b2 = (const float*)d_in[20];
  const float* att_W = (const float*)d_in[21];
  const float* att_b = (const float*)d_in[22];
  const float* pgru_Wk = (const float*)d_in[23];
  const float* pgru_Wr = (const float*)d_in[24];
  const float* pgru_b  = (const float*)d_in[25];
  const float* lgru_Wk = (const float*)d_in[26];
  const float* lgru_Wr = (const float*)d_in[27];
  const float* lgru_b  = (const float*)d_in[28];
  const float* ro_W1 = (const float*)d_in[29];
  const float* ro_b1 = (const float*)d_in[30];
  const float* ro_W2 = (const float*)d_in[31];
  const float* ro_b2 = (const float*)d_in[32];
  const float* ro_W3 = (const float*)d_in[33];
  const float* ro_b3 = (const float*)d_in[34];
  const int* flow_length  = (const int*)d_in[35];
  const int* link_to_path = (const int*)d_in[36];
  const int* path_to_link = (const int*)d_in[37];
  (void)flow_length; (void)in_sizes; (void)n_in; (void)out_size; (void)ws_size;

  // workspace layout (~32.6 MB, L2-resident)
  float* ps   = (float*)d_ws;                       // (NF,16)
  float* seqb = ps   + (size_t)NF * 16;             // (NF,9,16)
  float* ls   = seqb + (size_t)NF * 144;            // (NL,16)
  float* psm  = ls   + (size_t)NL * 16;             // (NL,16)

  flow_encode_kernel<<<(NF + 255)/256, 256, 0, stream>>>(
      flow_traffic, flow_packets, flow_pkts_pb, flow_bitrate_pb, flow_pkt_size,
      flow_type, flow_ipg_mean, ibg, flow_p90, rate, flow_ipg_var,
      flow_length, fe_W1, fe_b1, fe_W2, fe_b2, ps);
  link_encode_kernel<<<(NL + 255)/256, 256, 0, stream>>>(
      flow_traffic, link_capacity, max_link_load, path_to_link,
      le_W1, le_b1, le_W2, le_b2, ls);

  for (int it = 0; it < NITERS; ++it){
    path_gru_wmma_kernel<<<(NF + 127)/128, 256, 0, stream>>>(
        ls, ps, seqb, link_to_path, pgru_Wk, pgru_Wr, pgru_b);
    attention_wmma_kernel<<<NL/8, 256, 0, stream>>>(
        seqb, path_to_link, att_W, att_b, psm);
    link_gru_kernel<<<(NL + 255)/256, 256, 0, stream>>>(
        ls, psm, lgru_Wk, lgru_Wr, lgru_b);
  }

  readout_kernel<<<(NF + 255)/256, 256, 0, stream>>>(
      seqb, link_to_path, link_capacity,
      ro_W1, ro_b1, ro_W2, ro_b2, ro_W3, ro_b3, (float*)d_out);
}
